// SimpleRNN_42984032698625
// MI455X (gfx1250) — matrix operand order repaired and checked by reading, not compiled, not run on hardware
//
#include <hip/hip_runtime.h>
#include <math.h>

// ---------------------------------------------------------------------------
// SimpleRNN scan on MI455X (gfx1250): persistent grid, bf16 WMMA, LDS weights.
//   B=64, T=512, C=32, TH=32, H=1024, L=128.  K_total = 64 + 1024 = 1088.
//   WG g owns 16 hidden columns; wave w owns batch rows [16w,16w+16).
//   Per step/wave: 34 x v_wmma_f32_16x16x32_bf16.
// ---------------------------------------------------------------------------

typedef __attribute__((ext_vector_type(16))) __bf16        v16bf;
typedef __attribute__((ext_vector_type(8)))  float         v8f;
typedef __attribute__((ext_vector_type(8)))  unsigned int  v8u;

constexpr int B_   = 64;
constexpr int T_   = 512;
constexpr int C_   = 32;
constexpr int TH_  = 32;
constexpr int H_   = 1024;
constexpr int L_   = 128;
constexpr int KIN  = C_ + TH_;       // 64
constexpr int KC_RNN = (KIN + H_) / 32;  // 34 K-chunks of 32
constexpr int KC_H   = H_ / 32;          // 32
constexpr int KC_OUT = H_ / 32;          // 32
constexpr int NWG    = H_ / 16;          // 64 workgroups
constexpr int HSTATE = B_ * H_;          // bf16 elements per h buffer

static __device__ __forceinline__ unsigned short f2bf(float f) {
    unsigned u = __builtin_bit_cast(unsigned, f);
    u += 0x7FFFu + ((u >> 16) & 1u);      // round-to-nearest-even
    return (unsigned short)(u >> 16);
}
static __device__ __forceinline__ unsigned pack2(float lo, float hi) {
    return (unsigned)f2bf(lo) | ((unsigned)f2bf(hi) << 16);
}
static __device__ __forceinline__ v8f wmma_bf16(v8u a, v8u b, v8f c) {
    return __builtin_amdgcn_wmma_f32_16x16x32_bf16(
        false, __builtin_bit_cast(v16bf, a),
        false, __builtin_bit_cast(v16bf, b),
        (short)0, c, false, false);
}

__global__ void rnn_init_kernel(unsigned* sync, unsigned* hzero, int nwords) {
    const int i = blockIdx.x * blockDim.x + threadIdx.x;
    if (i == 0) *sync = 0u;
    for (int k = i; k < nwords; k += gridDim.x * blockDim.x) hzero[k] = 0u;
}

__global__ __launch_bounds__(128)
void rnn_persistent_kernel(const float* __restrict__ x,
                           const float* __restrict__ tf,
                           const float* __restrict__ Wrnn,
                           const float* __restrict__ brnn,
                           const float* __restrict__ Wout,
                           const float* __restrict__ bout,
                           float* __restrict__ out,
                           unsigned short* __restrict__ Hbuf,
                           unsigned* __restrict__ sync) {
    // B-fragments: [kc][lane] -> 8 dwords (two uint4). 34*32*32B = 34 KB.
    __shared__ uint4 ldsW[KC_RNN * 32 * 2];

    const int wg   = blockIdx.x;         // owns hidden cols [16*wg, 16*wg+16)
    const int lane = threadIdx.x & 31;
    const int wv   = threadIdx.x >> 5;   // wave id 0..3 -> M tile
    const int g    = lane >> 4;          // lane group (0|1)
    const int nloc = lane & 15;          // local column
    const int n0   = wg * 16;
    const int row0 = wv * 16;
    const int myrow = row0 + nloc;       // A-fragment row (batch index)

    // ---- Stage W_rnn[:, n0:n0+16] into LDS as bf16 B-fragments (once) ----
    // B frag (bf16 32x16): lane = N + 16*g; lane needs K = {8g..8g+7} and {8g+16..8g+23}.
    for (int kc = wv; kc < KC_RNN; kc += 4) {
        const int kbase = kc * 32 + g * 8;
        const float* wp = Wrnn + (size_t)kbase * H_ + (n0 + nloc);
        unsigned d[8];
#pragma unroll
        for (int i = 0; i < 8; ++i)
            d[i] = pack2(wp[(2 * i + (i >= 4 ? 8 : 0)) * H_], wp[(2 * i + 1 + (i >= 4 ? 8 : 0)) * H_]);
        ldsW[(kc * 32 + lane) * 2 + 0] = make_uint4(d[0], d[1], d[2], d[3]);
        ldsW[(kc * 32 + lane) * 2 + 1] = make_uint4(d[4], d[5], d[6], d[7]);
    }
    __syncthreads();

    const float biasc = brnn[n0 + nloc];

    for (int t = 0; t < T_; ++t) {
        const unsigned short* Hcur = Hbuf + (size_t)(t & 1) * HSTATE;
        unsigned short* Hnext = Hbuf + (size_t)((t + 1) & 1) * HSTATE;

        v8f acc = {};

        // ---- input features K=0..63 (f32 -> bf16 on the fly) ----
        // A frag (bf16 16x32): lane needs K = {8g..8g+7} and {8g+16..8g+23}.
        {
            const float* xp = x  + ((size_t)myrow * T_ + t) * C_;
            const float* tp = tf + ((size_t)myrow * T_ + t) * TH_;
#pragma unroll
            for (int kc = 0; kc < 2; ++kc) {
                const float* src = kc ? tp : xp;
                const float4 a0 = *(const float4*)(src + g * 8);
                const float4 a1 = *(const float4*)(src + g * 8 + 4);
                const float4 b0 = *(const float4*)(src + g * 8 + 16);
                const float4 b1 = *(const float4*)(src + g * 8 + 20);
                v8u au;
                au[0] = pack2(a0.x, a0.y); au[1] = pack2(a0.z, a0.w);
                au[2] = pack2(a1.x, a1.y); au[3] = pack2(a1.z, a1.w);
                au[4] = pack2(b0.x, b0.y); au[5] = pack2(b0.z, b0.w);
                au[6] = pack2(b1.x, b1.y); au[7] = pack2(b1.z, b1.w);
                const uint4 q0 = ldsW[(kc * 32 + lane) * 2 + 0];
                const uint4 q1 = ldsW[(kc * 32 + lane) * 2 + 1];
                v8u bu = {q0.x, q0.y, q0.z, q0.w, q1.x, q1.y, q1.z, q1.w};
                acc = wmma_bf16(au, bu, acc);
            }
        }

        // ---- recurrent part K=64..1087: h rows are bf16, fragment-direct ----
#pragma unroll 8
        for (int kc = 0; kc < KC_H; ++kc) {
            const unsigned short* hp =
                Hcur + (size_t)myrow * H_ + kc * 32 + g * 8;
            const uint4 h0 = *(const uint4*)hp;          // K {8g..8g+7}
            const uint4 h1 = *(const uint4*)(hp + 16);   // K {8g+16..8g+23}
            v8u au = {h0.x, h0.y, h0.z, h0.w, h1.x, h1.y, h1.z, h1.w};
            const int kb = kc + 2;
            const uint4 q0 = ldsW[(kb * 32 + lane) * 2 + 0];
            const uint4 q1 = ldsW[(kb * 32 + lane) * 2 + 1];
            v8u bu = {q0.x, q0.y, q0.z, q0.w, q1.x, q1.y, q1.z, q1.w};
            acc = wmma_bf16(au, bu, acc);
        }

        // ---- bias + tanh, publish next state (C/D: VGPR r -> M = 8g + r) ----
#pragma unroll
        for (int r = 0; r < 8; ++r) {
            const float hval = tanhf(acc[r] + biasc);
            const int row = row0 + g * 8 + r;
            Hnext[(size_t)row * H_ + n0 + nloc] = f2bf(hval);
        }

        // ---- grid-wide barrier: release -> arrive -> spin -> acquire ----
        __builtin_amdgcn_fence(__ATOMIC_RELEASE, "agent");   // per-wave global_wb
        __syncthreads();
        if (threadIdx.x == 0) {
            __hip_atomic_fetch_add(sync, 1u, __ATOMIC_RELAXED,
                                   __HIP_MEMORY_SCOPE_AGENT);
            const unsigned target = (unsigned)(t + 1) * NWG;
            while (__hip_atomic_load(sync, __ATOMIC_RELAXED,
                                     __HIP_MEMORY_SCOPE_AGENT) < target)
                __builtin_amdgcn_s_sleep(1);
        }
        __syncthreads();
        __builtin_amdgcn_fence(__ATOMIC_ACQUIRE, "agent");   // global_inv
    }

    // ---- final projection: out = h_last @ W_out + b_out (WGs 0..7) ----
    if (wg < 8) {
        __syncthreads();   // own waves done reading old LDS contents
        for (int kc = wv; kc < KC_OUT; kc += 4) {
            const int kbase = kc * 32 + g * 8;
            const float* wp = Wout + (size_t)kbase * L_ + (n0 + nloc);
            unsigned d[8];
#pragma unroll
            for (int i = 0; i < 8; ++i)
                d[i] = pack2(wp[(2 * i + (i >= 4 ? 8 : 0)) * L_], wp[(2 * i + 1 + (i >= 4 ? 8 : 0)) * L_]);
            ldsW[(kc * 32 + lane) * 2 + 0] = make_uint4(d[0], d[1], d[2], d[3]);
            ldsW[(kc * 32 + lane) * 2 + 1] = make_uint4(d[4], d[5], d[6], d[7]);
        }
        __syncthreads();

        const unsigned short* Hlast = Hbuf + (size_t)(T_ & 1) * HSTATE;
        v8f acc = {};
#pragma unroll 8
        for (int kc = 0; kc < KC_OUT; ++kc) {
            const unsigned short* hp =
                Hlast + (size_t)myrow * H_ + kc * 32 + g * 8;
            const uint4 h0 = *(const uint4*)hp;
            const uint4 h1 = *(const uint4*)(hp + 16);
            v8u au = {h0.x, h0.y, h0.z, h0.w, h1.x, h1.y, h1.z, h1.w};
            const uint4 q0 = ldsW[(kc * 32 + lane) * 2 + 0];
            const uint4 q1 = ldsW[(kc * 32 + lane) * 2 + 1];
            v8u bu = {q0.x, q0.y, q0.z, q0.w, q1.x, q1.y, q1.z, q1.w};
            acc = wmma_bf16(au, bu, acc);
        }
        const float bo = bout[n0 + nloc];
#pragma unroll
        for (int r = 0; r < 8; ++r) {
            const int row = row0 + g * 8 + r;
            out[(size_t)row * L_ + n0 + nloc] = acc[r] + bo;
        }
    }
}

extern "C" void kernel_launch(void* const* d_in, const int* in_sizes, int n_in,
                              void* d_out, int out_size, void* d_ws, size_t ws_size,
                              hipStream_t stream) {
    const float* x    = (const float*)d_in[0];   // [B,T,C]
    const float* tf   = (const float*)d_in[1];   // [B,T,TH]
    const float* Wrnn = (const float*)d_in[2];   // [1088,1024]
    const float* brnn = (const float*)d_in[3];   // [1024]
    const float* Wout = (const float*)d_in[4];   // [1024,128]
    const float* bout = (const float*)d_in[5];   // [128]
    float* out = (float*)d_out;                  // [B,1,L] = [64,128]

    unsigned char* ws = (unsigned char*)d_ws;
    unsigned* sync = (unsigned*)ws;                              // 256 B slot
    unsigned short* Hbuf = (unsigned short*)(ws + 256);          // 2x64x1024 bf16

    // Reset sync counter and zero h0 (buffer 0) every call.
    rnn_init_kernel<<<64, 256, 0, stream>>>(sync, (unsigned*)Hbuf, HSTATE / 2);

    rnn_persistent_kernel<<<NWG, 128, 0, stream>>>(
        x, tf, Wrnn, brnn, Wout, bout, out, Hbuf, sync);
}
